// SelfAttention_55018531062335
// MI455X (gfx1250) — compile-verified, matrix operand order checked by reading, not run on hardware
//
#include <hip/hip_runtime.h>

typedef __attribute__((ext_vector_type(16))) __bf16 bf16x16;
typedef __attribute__((ext_vector_type(8)))  __bf16 bf16x8;
typedef __attribute__((ext_vector_type(4)))  __bf16 bf16x4;
typedef __attribute__((ext_vector_type(8)))  float  floatx8;
typedef __attribute__((ext_vector_type(4)))  float  floatx4;

#define WMMA_BF16(a, b, c) \
  __builtin_amdgcn_wmma_f32_16x16x32_bf16(false, (a), false, (b), (short)0, (c), false, false)

// ---- CDNA5 async global->LDS copy (ASYNCcnt path, CDNA5 ISA) ----
// GV mode: per-lane 64-bit global address, per-lane 32-bit LDS offset in VDST.
// Generic pointers into LDS carry the wave-relative offset in bits [31:0].
static __device__ inline void async_ld16(const void* gptr, void* lptr) {
  unsigned l = (unsigned)(unsigned long long)lptr;
  asm volatile("global_load_async_to_lds_b128 %0, %1, off"
               :: "v"(l), "v"(gptr) : "memory");
}
static __device__ inline void wait_async_le4() {
  asm volatile("s_wait_asynccnt 0x4" ::: "memory");
}
static __device__ inline void wait_async_0() {
  asm volatile("s_wait_asynccnt 0x0" ::: "memory");
}

// A/B operand fragment for v_wmma_f32_16x16x32_bf16, per ISA 16-bit layout:
// lane (r, hl) holds row r, K = base + hl*8 + i (elems 0..7) and base + 16 + hl*8 + i (elems 8..15)
static __device__ inline bf16x16 frag_contig(const __bf16* p) {
  bf16x16 f;
#pragma unroll
  for (int i = 0; i < 8; ++i) { f[i] = p[i]; f[i + 8] = p[16 + i]; }
  return f;
}
static __device__ inline bf16x16 frag_strided(const __bf16* p, int stride) {
  bf16x16 f;
#pragma unroll
  for (int i = 0; i < 8; ++i) {
    f[i]     = p[(size_t)i * stride];
    f[i + 8] = p[(size_t)(16 + i) * stride];
  }
  return f;
}

__global__ __launch_bounds__(256)
void cvt_f32_bf16(const float* __restrict__ in, __bf16* __restrict__ out, int n) {
  int i = blockIdx.x * 256 + threadIdx.x;
  if (i < n) out[i] = (__bf16)in[i];
}

// C[m,n] = sum_k A[m,k] * W[n,k] + bias[n]   (torch Linear: x @ W.T + b)
// A: M x K row-major bf16, W: N x K row-major bf16 (both K-contiguous -> TN GEMM)
// Double-buffered LDS fed by global_load_async_to_lds_b128.
template <bool OUT_F32>
__global__ __launch_bounds__(256)
void gemm_tn(const __bf16* __restrict__ A, const __bf16* __restrict__ W,
             const float* __restrict__ bias, void* __restrict__ Cout,
             int M, int N, int K) {
  constexpr int BM = 128, BN = 128, BK = 32, PAD = 8, LDR = BK + PAD;
  __shared__ __bf16 sA[2][BM][LDR];
  __shared__ __bf16 sB[2][BN][LDR];

  const int tid  = threadIdx.x;
  const int wave = tid >> 5;
  const int lane = tid & 31;
  const int hl   = lane >> 4;   // which 16-lane half of the wave
  const int r    = lane & 15;

  const int tileM0 = blockIdx.y * BM;
  const int tileN0 = blockIdx.x * BN;
  const int waveM  = wave & 1;   // 2 waves along M (64 rows each)
  const int waveN  = wave >> 1;  // 4 waves along N (32 cols each)

  // Per-thread copy slots: 512 chunks of 8 bf16 per matrix tile, 2 per thread.
  const int row0 = tid >> 2,          col0 = (tid & 3) << 3;
  const int row1 = (tid + 256) >> 2,  col1 = ((tid + 256) & 3) << 3;

  floatx8 acc[4][2] = {};

  // Prologue: fill buffer 0 (4 async instructions per thread)
  {
    async_ld16(A + (size_t)(tileM0 + row0) * K + col0, &sA[0][row0][col0]);
    async_ld16(W + (size_t)(tileN0 + row0) * K + col0, &sB[0][row0][col0]);
    async_ld16(A + (size_t)(tileM0 + row1) * K + col1, &sA[0][row1][col1]);
    async_ld16(W + (size_t)(tileN0 + row1) * K + col1, &sB[0][row1][col1]);
  }

  int buf = 0;
  for (int k0 = 0; k0 < K; k0 += BK) {
    const bool more = (k0 + BK) < K;
    if (more) {  // prefetch next tile into the other buffer
      const int kn = k0 + BK, nb = buf ^ 1;
      async_ld16(A + (size_t)(tileM0 + row0) * K + kn + col0, &sA[nb][row0][col0]);
      async_ld16(W + (size_t)(tileN0 + row0) * K + kn + col0, &sB[nb][row0][col0]);
      async_ld16(A + (size_t)(tileM0 + row1) * K + kn + col1, &sA[nb][row1][col1]);
      async_ld16(W + (size_t)(tileN0 + row1) * K + kn + col1, &sB[nb][row1][col1]);
      wait_async_le4();   // async loads complete in order: current buf is done
    } else {
      wait_async_0();
    }
    __syncthreads();

    bf16x16 af[4], bfr[2];
#pragma unroll
    for (int mi = 0; mi < 4; ++mi)
      af[mi] = frag_contig(&sA[buf][waveM * 64 + mi * 16 + r][hl * 8]);
#pragma unroll
    for (int ni = 0; ni < 2; ++ni)
      bfr[ni] = frag_contig(&sB[buf][waveN * 32 + ni * 16 + r][hl * 8]);
#pragma unroll
    for (int mi = 0; mi < 4; ++mi)
#pragma unroll
      for (int ni = 0; ni < 2; ++ni)
        acc[mi][ni] = WMMA_BF16(af[mi], bfr[ni], acc[mi][ni]);

    __syncthreads();
    buf ^= 1;
  }

  // Epilogue: C/D layout -> VGPR j: row = j + hl*8, col = r (within 16x16 tile)
#pragma unroll
  for (int mi = 0; mi < 4; ++mi) {
#pragma unroll
    for (int ni = 0; ni < 2; ++ni) {
      int col  = tileN0 + waveN * 32 + ni * 16 + r;
      float bb = bias[col];
#pragma unroll
      for (int j = 0; j < 8; ++j) {
        int row  = tileM0 + waveM * 64 + mi * 16 + j + hl * 8;
        float v  = acc[mi][ni][j] + bb;
        if (OUT_F32) ((float*)Cout)[(size_t)row * N + col] = v;
        else        ((__bf16*)Cout)[(size_t)row * N + col] = (__bf16)v;
      }
    }
  }
}

// Fused per (b, h, 16-row q-tile): scores (WMMA) -> softmax (LDS) -> attn write -> ctx = P @ V (WMMA)
__global__ __launch_bounds__(256)
void attn_fused(const __bf16* __restrict__ Qb, const __bf16* __restrict__ Kb,
                const __bf16* __restrict__ Vb, float* __restrict__ attn,
                __bf16* __restrict__ ctx) {
  constexpr int S = 2048, D = 1024, H = 16, HD = 64;
  __shared__ __bf16 scb[16][S];  // exactly 64 KB

  const int tid  = threadIdx.x;
  const int wave = tid >> 5;
  const int lane = tid & 31;
  const int hl   = lane >> 4;
  const int r    = lane & 15;

  const int q0 = blockIdx.x * 16;
  const int h  = blockIdx.y;
  const int b  = blockIdx.z;

  // ---- Phase 1: scores = (Q K^T) / sqrt(HD), tile-by-tile into LDS (bf16) ----
  const size_t qrow = (size_t)(b * S + q0 + r) * D + h * HD;
  bf16x16 aq0 = frag_contig(Qb + qrow + hl * 8);        // K-dim 0..31 of head
  bf16x16 aq1 = frag_contig(Qb + qrow + 32 + hl * 8);   // K-dim 32..63

  for (int ct = wave; ct < S / 16; ct += 8) {
    const __bf16* kp = Kb + (size_t)(b * S + ct * 16 + r) * D + h * HD + hl * 8;
    bf16x16 bk0 = frag_contig(kp);
    bf16x16 bk1 = frag_contig(kp + 32);
    floatx8 c0 = {}, c1 = {};           // independent accumulators: no D->C chain
    c0 = WMMA_BF16(aq0, bk0, c0);
    c1 = WMMA_BF16(aq1, bk1, c1);
    floatx8 cs = c0 + c1;
#pragma unroll
    for (int j = 0; j < 8; ++j)
      scb[j + hl * 8][ct * 16 + r] = (__bf16)(cs[j] * 0.125f);  // 1/sqrt(64)
  }
  __syncthreads();

  // ---- Phase 2: row softmax; each 16-lane half owns one row.
  // Thread (row, c0) owns the contiguous 128-element segment [c0*128, c0*128+128)
  // so LDS traffic vectorizes to ds_load_b128 / ds_store_b128.
  {
    int row = tid >> 4;   // 0..15
    int c0  = tid & 15;
    __bf16* prow = &scb[row][c0 * 128];

    float m = -1e30f;
#pragma unroll
    for (int i = 0; i < 16; ++i) {
      bf16x8 v = *(bf16x8*)(prow + i * 8);
#pragma unroll
      for (int e = 0; e < 8; ++e) m = fmaxf(m, (float)v[e]);
    }
#pragma unroll
    for (int off = 8; off; off >>= 1) m = fmaxf(m, __shfl_xor(m, off, 16));

    float s = 0.f;
#pragma unroll
    for (int i = 0; i < 16; ++i) {
      bf16x8 v = *(bf16x8*)(prow + i * 8);
      bf16x8 o;
#pragma unroll
      for (int e = 0; e < 8; ++e) {
        float ev = __expf((float)v[e] - m);
        s += ev;
        o[e] = (__bf16)ev;
      }
      *(bf16x8*)(prow + i * 8) = o;   // same-wave LDS ops stay in order
    }
#pragma unroll
    for (int off = 8; off; off >>= 1) s += __shfl_xor(s, off, 16);
    float inv  = 1.f / s;
    float inv0 = __shfl(inv, 0, 32);    // row 2*wave
    float inv1 = __shfl(inv, 16, 32);   // row 2*wave + 1

    // Normalize + write attn: wave owns rows {2w, 2w+1}; lane handles 4
    // contiguous elements -> float4 global stores, coalesced per wave.
    int rbase = wave * 2;
    float* ab = attn + ((size_t)(b * H + h) * S + q0 + rbase) * S;
#pragma unroll
    for (int rr = 0; rr < 2; ++rr) {
      float invr = (rr == 0) ? inv0 : inv1;
      for (int c = lane * 4; c < S; c += 128) {
        bf16x4 v = *(bf16x4*)&scb[rbase + rr][c];
        floatx4 p;
        bf16x4 o;
#pragma unroll
        for (int e = 0; e < 4; ++e) {
          p[e] = (float)v[e] * invr;
          o[e] = (__bf16)p[e];
        }
        *(floatx4*)&ab[(size_t)rr * S + c] = p;
        *(bf16x4*)&scb[rbase + rr][c] = o;
      }
    }
  }
  __syncthreads();

  // ---- Phase 3: ctx(16 x 64) = P(16 x S) @ V(S x 64); one wave per 16-col tile.
  // Two interleaved accumulators so consecutive WMMAs are independent.
  if (wave < 4) {
    int nt = wave;
    floatx8 acc0 = {}, acc1 = {};
    for (int kb = 0; kb < S; kb += 64) {
      bf16x16 a0 = frag_contig(&scb[r][kb + hl * 8]);
      const __bf16* pv0 = Vb + (size_t)(b * S + kb + hl * 8) * D + h * HD + nt * 16 + r;
      bf16x16 bv0 = frag_strided(pv0, D);
      acc0 = WMMA_BF16(a0, bv0, acc0);

      bf16x16 a1 = frag_contig(&scb[r][kb + 32 + hl * 8]);
      const __bf16* pv1 = Vb + (size_t)(b * S + kb + 32 + hl * 8) * D + h * HD + nt * 16 + r;
      bf16x16 bv1 = frag_strided(pv1, D);
      acc1 = WMMA_BF16(a1, bv1, acc1);
    }
    floatx8 accs = acc0 + acc1;
#pragma unroll
    for (int j = 0; j < 8; ++j)
      ctx[(size_t)(b * S + q0 + j + hl * 8) * D + h * HD + nt * 16 + r] = (__bf16)accs[j];
  }
}

extern "C" void kernel_launch(void* const* d_in, const int* in_sizes, int n_in,
                              void* d_out, int out_size, void* d_ws, size_t ws_size,
                              hipStream_t stream) {
  (void)in_sizes; (void)n_in; (void)out_size; (void)ws_size;
  const int B = 4, S = 2048, D = 1024, H = 16;
  const int M = B * S;  // 8192

  const float* x  = (const float*)d_in[0];
  const float* Wq = (const float*)d_in[1];
  const float* bq = (const float*)d_in[2];
  const float* Wk = (const float*)d_in[3];
  const float* bk = (const float*)d_in[4];
  const float* Wv = (const float*)d_in[5];
  const float* bv = (const float*)d_in[6];
  const float* Wo = (const float*)d_in[7];
  const float* bo = (const float*)d_in[8];

  char* ws = (char*)d_ws;
  size_t off = 0;
  auto alloc = [&](size_t bytes) -> void* {
    void* p = ws + off;
    off += (bytes + 255) & ~(size_t)255;
    return p;
  };
  __bf16* xb  = (__bf16*)alloc((size_t)M * D * 2);
  __bf16* Wqb = (__bf16*)alloc((size_t)D * D * 2);
  __bf16* Wkb = (__bf16*)alloc((size_t)D * D * 2);
  __bf16* Wvb = (__bf16*)alloc((size_t)D * D * 2);
  __bf16* Wob = (__bf16*)alloc((size_t)D * D * 2);
  __bf16* Qb  = (__bf16*)alloc((size_t)M * D * 2);
  __bf16* Kb  = (__bf16*)alloc((size_t)M * D * 2);
  __bf16* Vb  = (__bf16*)alloc((size_t)M * D * 2);
  __bf16* Cb  = (__bf16*)alloc((size_t)M * D * 2);

  const int nx = M * D;   // 8,388,608
  const int nw = D * D;   // 1,048,576
  cvt_f32_bf16<<<(nx + 255) / 256, 256, 0, stream>>>(x,  xb,  nx);
  cvt_f32_bf16<<<(nw + 255) / 256, 256, 0, stream>>>(Wq, Wqb, nw);
  cvt_f32_bf16<<<(nw + 255) / 256, 256, 0, stream>>>(Wk, Wkb, nw);
  cvt_f32_bf16<<<(nw + 255) / 256, 256, 0, stream>>>(Wv, Wvb, nw);
  cvt_f32_bf16<<<(nw + 255) / 256, 256, 0, stream>>>(Wo, Wob, nw);

  dim3 gg(D / 128, M / 128);  // (8, 64)
  gemm_tn<false><<<gg, 256, 0, stream>>>(xb, Wqb, bq, Qb, M, D, D);
  gemm_tn<false><<<gg, 256, 0, stream>>>(xb, Wkb, bk, Kb, M, D, D);
  gemm_tn<false><<<gg, 256, 0, stream>>>(xb, Wvb, bv, Vb, M, D, D);

  float* outp  = (float*)d_out;
  float* attnp = outp + (size_t)M * D;
  attn_fused<<<dim3(S / 16, H, B), 256, 0, stream>>>(Qb, Kb, Vb, attnp, Cb);

  gemm_tn<true><<<gg, 256, 0, stream>>>(Cb, Wob, bo, outp, M, D, D);
}
